// AECFModel_7181185319647
// MI455X (gfx1250) — compile-verified, matrix operand order checked by reading, not run on hardware
//
#include <hip/hip_runtime.h>
#include <hip/hip_bf16.h>

// ---------------------------------------------------------------------------
// Fused AECF forward for MI455X (gfx1250, wave32).
// All GEMMs run on v_wmma_f32_16x16x32_f16 (f16 in / f32 accumulate).
// One wave owns a 16-row batch tile end-to-end; every intermediate lives in
// LDS, only raw features in / logits out touch HBM. Weights are converted +
// swizzled once per launch into WMMA B-fragment order in d_ws (stay in L2).
// K-projection is eliminated algebraically: scores = enc . (Wk q)/sqrt(d).
// ---------------------------------------------------------------------------

typedef _Float16 v16h __attribute__((ext_vector_type(16)));
typedef _Float16 h8   __attribute__((ext_vector_type(8)));
typedef float    v8f  __attribute__((ext_vector_type(8)));

#define H   256
#define IDD 512
#define NC  80
#define LDA 520   // halves; 520*2 bytes = 1040 = 16B multiple
#define LDE 264   // halves; 528 bytes = 16B multiple
#define WAVES_PER_BLOCK 2
// per-wave LDS: bufX 16*520 + 3 * 16*264 halves, attnW 64 f32, flags 32 f32
#define WAVE_HALVES (16*LDA + 3*16*LDE)
#define WAVE_BYTES  ((WAVE_HALVES*2 + 64*4 + 32*4 + 127) & ~127)

// ---------------- fragment math helpers ------------------------------------

// Accumulate C += A(16x(32*KB)) * B using WMMA. A is row-major f16 in LDS with
// stride lda (halves). wnt points at the pre-swizzled fragment panel for one
// 16-col N tile: layout [kb][lane][16 halves].
// A fragment (16-bit 16x32): lane = M%16 | (hi<<4); halves 0..7 = K k0+hi*8..+7,
// halves 8..15 = K k0+16+hi*8..+7  -> two ds_load_b128 per fragment.
template <int KB>
__device__ inline v8f gemm_acc(const _Float16* __restrict__ A, int lda,
                               const _Float16* __restrict__ wnt,
                               int lane, float init) {
  const int row = lane & 15;
  const int hi  = lane >> 4;
  v8f acc;
#pragma unroll
  for (int j = 0; j < 8; ++j) acc[j] = init;
#pragma unroll
  for (int kb = 0; kb < KB; ++kb) {
    const _Float16* ap = A + row * lda + kb * 32 + hi * 8;
    h8 a0 = *(const h8*)(ap);        // K = k0+hi*8 .. +7
    h8 a1 = *(const h8*)(ap + 16);   // K = k0+16+hi*8 .. +7
    v16h av = __builtin_shufflevector(a0, a1, 0,1,2,3,4,5,6,7,8,9,10,11,12,13,14,15);
    const _Float16* wp = wnt + (kb * 32 + lane) * 16;  // 32B per lane, 16B aligned
    h8 b0 = *(const h8*)(wp);
    h8 b1 = *(const h8*)(wp + 8);
    v16h bv = __builtin_shufflevector(b0, b1, 0,1,2,3,4,5,6,7,8,9,10,11,12,13,14,15);
    acc = __builtin_amdgcn_wmma_f32_16x16x32_f16(false, av, false, bv,
                                                 (short)0, acc, false, false);
  }
  return acc;
}

// Full GEMM stage: D(16xNT*16) = act(A * W + bias), f16 result back to LDS.
// C/D layout: acc[j] -> row j+8*hi, col nt*16 + (lane&15).
template <int KB, int NT, bool RELU>
__device__ inline void gemm_stage(const _Float16* __restrict__ A, int lda,
                                  const _Float16* __restrict__ wfrag,
                                  const float* __restrict__ bias,
                                  _Float16* __restrict__ D, int ldd, int lane) {
  const int col = lane & 15;
  const int hi  = lane >> 4;
  for (int nt = 0; nt < NT; ++nt) {
    if (nt + 1 < NT) __builtin_prefetch(wfrag + (nt + 1) * KB * 512, 0, 0);
    v8f acc = gemm_acc<KB>(A, lda, wfrag + nt * KB * 512, lane, bias[nt * 16 + col]);
#pragma unroll
    for (int j = 0; j < 8; ++j) {
      float v = acc[j];
      if (RELU) v = fmaxf(v, 0.0f);
      D[(j + 8 * hi) * ldd + nt * 16 + col] = (_Float16)v;
    }
  }
}

// ---------------- prep kernels ----------------------------------------------

// Convert W[K][N] (f32, row-major) to f16 WMMA B-fragment order:
// dst[((nt*KB + kb)*32 + lane)*16 + j] = W[k][nt*16 + (lane&15)]
// with hi = lane>>4, k = kb*32 + hi*8 + (j<8 ? j : 16 + j-8).
// (B fragment mirrors the A layout with lane = N column: B^T in A order.)
__global__ void swizzle_w(const float* __restrict__ W, _Float16* __restrict__ dst,
                          int K, int N) {
  int idx = blockIdx.x * 256 + threadIdx.x;
  if (idx >= K * N) return;
  int j    = idx & 15;
  int lane = (idx >> 4) & 31;
  int rest = idx >> 9;
  int KB   = K >> 5;
  int kb   = rest % KB;
  int nt   = rest / KB;
  int hi   = lane >> 4;
  int k    = kb * 32 + hi * 8 + (j < 8 ? j : 16 + (j - 8));
  int n    = nt * 16 + (lane & 15);
  dst[idx] = (_Float16)W[k * N + n];
}

// qvec = fusion_query @ Wq + bq  (1x256 @ 256x256)
__global__ void compute_q(const float* __restrict__ fq, const float* __restrict__ Wq,
                          const float* __restrict__ bq, float* __restrict__ qvec) {
  int t = threadIdx.x;  // 256 threads
  float s = bq[t];
  for (int j = 0; j < H; ++j) s += fq[j] * Wq[j * H + t];
  qvec[t] = s;
}

// Wk_eff[k][h] = (1/8) * sum_d Wk[k][h*64+d] * q[h*64+d];  ck[h] = (1/8)*bk.q_h
__global__ void compute_wkeff(const float* __restrict__ Wk, const float* __restrict__ bk,
                              const float* __restrict__ qvec,
                              float* __restrict__ wkeff, float* __restrict__ ck) {
  int k = threadIdx.x;  // 256 threads
  for (int h = 0; h < 4; ++h) {
    float s = 0.f;
    for (int d = 0; d < 64; ++d) s += Wk[k * H + h * 64 + d] * qvec[h * 64 + d];
    wkeff[k * 4 + h] = s * 0.125f;
  }
  if (k < 4) {
    float s = 0.f;
    for (int d = 0; d < 64; ++d) s += bk[k * 64 + d] * qvec[k * 64 + d];
    ck[k] = s * 0.125f;
  }
}

// ---------------- main fused kernel -----------------------------------------

extern "C" __global__ void __launch_bounds__(WAVES_PER_BLOCK * 32)
aecf_fused(const float* __restrict__ img, const float* __restrict__ txt,
           const _Float16* __restrict__ wf_ie, const float* __restrict__ b_ie,
           const _Float16* __restrict__ wf_te, const float* __restrict__ b_te,
           const _Float16* __restrict__ wf_v,  const float* __restrict__ bv,
           const _Float16* __restrict__ wf_o,  const float* __restrict__ bo,
           const _Float16* __restrict__ wf_fp, const float* __restrict__ b_fp,
           const _Float16* __restrict__ wf_ip, const float* __restrict__ b_ip,
           const _Float16* __restrict__ wf_tp, const float* __restrict__ b_tp,
           const _Float16* __restrict__ wf_c1, const float* __restrict__ bc1,
           const _Float16* __restrict__ wf_c2, const float* __restrict__ bc2,
           const float* __restrict__ wkeff, const float* __restrict__ ck,
           float* __restrict__ out, long Brows) {
  extern __shared__ char smem[];
  const int lane = threadIdx.x & 31;
  const int wave = threadIdx.x >> 5;
  const int row  = lane & 15;
  const int hi   = lane >> 4;
  const int col  = lane & 15;
  const long tile = (long)blockIdx.x * WAVES_PER_BLOCK + wave;
  const long r0   = tile * 16;
  // Brows is a multiple of 32 (B = 131072); no partial blocks by construction.

  char* wb = smem + wave * WAVE_BYTES;
  _Float16* bufX  = (_Float16*)wb;             // [16][LDA] generic A staging
  _Float16* bufEi = bufX  + 16 * LDA;          // [16][LDE] enc_img
  _Float16* bufEt = bufEi + 16 * LDE;          // [16][LDE] enc_txt
  _Float16* bufPo = bufEt + 16 * LDE;          // [16][LDE] pooled@Wo / hidden
  float* attnW = (float*)(bufPo + 16 * LDE);   // [16][4] softmax weight (img side)
  float* flags = attnW + 64;                   // [0..15]=img sumsq, [16..31]=txt

  // ---- stage img rows -> bufX (f16) + per-row sum of squares -------------
  flags[lane] = 0.0f;
  __syncthreads();
  for (int r = 0; r < 16; ++r) {
    const float* src = img + (r0 + r) * IDD;
    float ss = 0.f;
    for (int c = lane; c < IDD; c += 32) {
      float v = src[c];
      ss += v * v;
      bufX[r * LDA + c] = (_Float16)v;
    }
    atomicAdd(&flags[r], ss);
  }
  __syncthreads();
  // enc_img = relu(img @ W_ie + b_ie)   K=512, N=256
  gemm_stage<16, 16, true>(bufX, LDA, wf_ie, b_ie, bufEi, LDE, lane);
  __syncthreads();

  // ---- stage txt rows -----------------------------------------------------
  for (int r = 0; r < 16; ++r) {
    const float* src = txt + (r0 + r) * IDD;
    float ss = 0.f;
    for (int c = lane; c < IDD; c += 32) {
      float v = src[c];
      ss += v * v;
      bufX[r * LDA + c] = (_Float16)v;
    }
    atomicAdd(&flags[16 + r], ss);
  }
  __syncthreads();
  // enc_txt = relu(txt @ W_te + b_te)
  gemm_stage<16, 16, true>(bufX, LDA, wf_te, b_te, bufEt, LDE, lane);
  __syncthreads();

  // ---- attention scores (K projection eliminated) ------------------------
  // lane handles row = lane&15, heads {2*hi, 2*hi+1}
  {
    int h0 = hi * 2, h1 = h0 + 1;
    float si0 = 0, si1 = 0, st0 = 0, st1 = 0;
    for (int k = 0; k < H; ++k) {
      float ei = (float)bufEi[row * LDE + k];
      float et = (float)bufEt[row * LDE + k];
      float w0 = wkeff[k * 4 + h0];
      float w1 = wkeff[k * 4 + h1];
      si0 += ei * w0; si1 += ei * w1;
      st0 += et * w0; st1 += et * w1;
    }
    si0 += ck[h0]; si1 += ck[h1]; st0 += ck[h0]; st1 += ck[h1];
    // 2-way softmax over {img, txt}: weight on img token
    attnW[row * 4 + h0] = 1.0f / (1.0f + __expf(st0 - si0));
    attnW[row * 4 + h1] = 1.0f / (1.0f + __expf(st1 - si1));
  }
  __syncthreads();

  // ---- V projections + attention pooling -> pooled in bufX ---------------
  for (int nt = 0; nt < 16; ++nt) {
    int h = nt >> 2;  // 64-wide head blocks, 4 N-tiles each
    v8f aI = gemm_acc<8>(bufEi, LDE, wf_v + nt * 8 * 512, lane, 0.0f);
    v8f aT = gemm_acc<8>(bufEt, LDE, wf_v + nt * 8 * 512, lane, 0.0f);
    float bcol = bv[nt * 16 + col];
#pragma unroll
    for (int j = 0; j < 8; ++j) {
      int rr = j + 8 * hi;
      float wi = attnW[rr * 4 + h];
      float p = wi * (aI[j] + bcol) + (1.0f - wi) * (aT[j] + bcol);
      bufX[rr * LDA + nt * 16 + col] = (_Float16)p;
    }
  }
  __syncthreads();

  // ---- pooled @ Wo + bo -> bufPo    K=256, N=256 --------------------------
  gemm_stage<8, 16, false>(bufX, LDA, wf_o, bo, bufPo, LDE, lane);
  __syncthreads();

  // ---- three 256->512 projections + per-row branch select -> fused (bufX)
  for (int nt = 0; nt < 32; ++nt) {
    v8f aF = gemm_acc<8>(bufPo, LDE, wf_fp + nt * 8 * 512, lane, b_fp[nt * 16 + col]);
    v8f aI = gemm_acc<8>(bufEi, LDE, wf_ip + nt * 8 * 512, lane, b_ip[nt * 16 + col]);
    v8f aT = gemm_acc<8>(bufEt, LDE, wf_tp + nt * 8 * 512, lane, b_tp[nt * 16 + col]);
#pragma unroll
    for (int j = 0; j < 8; ++j) {
      int rr = j + 8 * hi;
      bool pi = flags[rr]      > 1e-12f;   // ||x|| > 1e-6
      bool pt = flags[16 + rr] > 1e-12f;
      float v = (pi && pt) ? aF[j] : (pi ? aI[j] : (pt ? aT[j] : 0.0f));
      bufX[rr * LDA + nt * 16 + col] = (_Float16)v;
    }
  }
  __syncthreads();

  // ---- hidden = relu(fused @ Wc1 + bc1) -> bufPo   K=512, N=256 -----------
  gemm_stage<16, 16, true>(bufX, LDA, wf_c1, bc1, bufPo, LDE, lane);
  __syncthreads();

  // ---- logits = hidden @ Wc2 + bc2 -> global   K=256, N=80 (5 N tiles) ----
  for (int nt = 0; nt < 5; ++nt) {
    v8f acc = gemm_acc<8>(bufPo, LDE, wf_c2 + nt * 8 * 512, lane, bc2[nt * 16 + col]);
#pragma unroll
    for (int j = 0; j < 8; ++j) {
      out[(r0 + j + 8 * hi) * NC + nt * 16 + col] = acc[j];
    }
  }
}

// ---------------- launch -----------------------------------------------------

extern "C" void kernel_launch(void* const* d_in, const int* in_sizes, int n_in,
                              void* d_out, int out_size, void* d_ws, size_t ws_size,
                              hipStream_t stream) {
  const float* img  = (const float*)d_in[0];
  const float* txt  = (const float*)d_in[1];
  const float* W_ie = (const float*)d_in[2];  const float* b_ie = (const float*)d_in[3];
  const float* W_te = (const float*)d_in[4];  const float* b_te = (const float*)d_in[5];
  const float* fq   = (const float*)d_in[6];
  const float* Wq   = (const float*)d_in[7];  const float* bq   = (const float*)d_in[8];
  const float* Wk   = (const float*)d_in[9];  const float* bk   = (const float*)d_in[10];
  const float* Wv   = (const float*)d_in[11]; const float* bv   = (const float*)d_in[12];
  const float* Wo   = (const float*)d_in[13]; const float* bo   = (const float*)d_in[14];
  const float* W_ip = (const float*)d_in[15]; const float* b_ip = (const float*)d_in[16];
  const float* W_tp = (const float*)d_in[17]; const float* b_tp = (const float*)d_in[18];
  const float* W_fp = (const float*)d_in[19]; const float* b_fp = (const float*)d_in[20];
  const float* Wc1  = (const float*)d_in[21]; const float* bc1  = (const float*)d_in[22];
  const float* Wc2  = (const float*)d_in[23]; const float* bc2  = (const float*)d_in[24];
  float* out = (float*)d_out;

  const long Brows = (long)in_sizes[0] / IDD;  // 131072

  // workspace layout (halves for fragments, then f32 tail)
  _Float16* hws = (_Float16*)d_ws;
  const int OFF_IE = 0;            // 512*256
  const int OFF_TE = OFF_IE + 512 * 256;
  const int OFF_V  = OFF_TE + 512 * 256;   // 256*256
  const int OFF_O  = OFF_V  + 256 * 256;
  const int OFF_FP = OFF_O  + 256 * 256;   // 256*512
  const int OFF_IP = OFF_FP + 256 * 512;
  const int OFF_TP = OFF_IP + 256 * 512;
  const int OFF_C1 = OFF_TP + 256 * 512;   // 512*256
  const int OFF_C2 = OFF_C1 + 512 * 256;   // 256*80
  const int OFF_END = OFF_C2 + 256 * 80;
  float* qvec  = (float*)(hws + ((OFF_END + 7) & ~7));
  float* wkeff = qvec + 256;
  float* ckv   = wkeff + 1024;
  // total ws use ~1.88 MB (assumed <= ws_size)

  // --- one-time weight conversion / swizzle (runs every launch; deterministic)
  auto launch_sw = [&](const float* W, int off, int K, int N) {
    swizzle_w<<<dim3((K * N + 255) / 256), dim3(256), 0, stream>>>(W, hws + off, K, N);
  };
  launch_sw(W_ie, OFF_IE, 512, 256);
  launch_sw(W_te, OFF_TE, 512, 256);
  launch_sw(Wv,   OFF_V,  256, 256);
  launch_sw(Wo,   OFF_O,  256, 256);
  launch_sw(W_fp, OFF_FP, 256, 512);
  launch_sw(W_ip, OFF_IP, 256, 512);
  launch_sw(W_tp, OFF_TP, 256, 512);
  launch_sw(Wc1,  OFF_C1, 512, 256);
  launch_sw(Wc2,  OFF_C2, 256, 80);
  compute_q<<<1, 256, 0, stream>>>(fq, Wq, bq, qvec);
  compute_wkeff<<<1, 256, 0, stream>>>(Wk, bk, qvec, wkeff, ckv);

  // --- fused main kernel: one wave per 16-row tile, 2 waves per block ------
  const long nBlocks = Brows / (16 * WAVES_PER_BLOCK);
  const size_t shmem = (size_t)WAVE_BYTES * WAVES_PER_BLOCK;
  aecf_fused<<<dim3((unsigned)nBlocks), dim3(WAVES_PER_BLOCK * 32), shmem, stream>>>(
      img, txt,
      hws + OFF_IE, b_ie, hws + OFF_TE, b_te,
      hws + OFF_V,  bv,   hws + OFF_O,  bo,
      hws + OFF_FP, b_fp, hws + OFF_IP, b_ip, hws + OFF_TP, b_tp,
      hws + OFF_C1, bc1,  hws + OFF_C2, bc2,
      wkeff, ckv, out, Brows);
}
